// Decoder_30623116820881
// MI455X (gfx1250) — matrix operand order repaired and checked by reading, not compiled, not run on hardware
//
#include <hip/hip_runtime.h>

// ---- problem constants ----
#define B   32
#define T   32
#define S   100
#define H   512
#define E   300
#define V   45000
#define OOV 10
#define VEXT 45010
#define NPAD 45056          // 704 * 64, padded head columns
#define NEG_INF (-1.0e12f)

typedef __attribute__((ext_vector_type(16))) __bf16        v16bf;
typedef __attribute__((ext_vector_type(8)))  float         v8f;
typedef __attribute__((ext_vector_type(4)))  unsigned int  u32x4;

union Frag { u32x4 q[2]; v16bf v; };   // 32 bytes = 16 bf16 per lane

__device__ __forceinline__ unsigned short f2bf(float f) {
    unsigned int u = __float_as_uint(f);
    unsigned int r = (u + 0x7FFFu + ((u >> 16) & 1u)) >> 16;   // RNE
    return (unsigned short)r;
}

// ---------------- one-time: W_out [H,V] f32 -> Wt [NPAD,H] bf16 (transposed) --------
__global__ void wt_transpose_kernel(const float* __restrict__ W_out,
                                    unsigned short* __restrict__ Wt) {
    __shared__ float tile[32][33];
    int nx = blockIdx.x, ky = blockIdx.y;
    int tx = threadIdx.x, ty = threadIdx.y;            // block (32,8)
    #pragma unroll
    for (int j = 0; j < 4; ++j) {
        int k = ky * 32 + ty + j * 8;
        int n = nx * 32 + tx;
        tile[ty + j * 8][tx] = (n < V) ? W_out[(long)k * V + n] : 0.0f;
    }
    __syncthreads();
    #pragma unroll
    for (int j = 0; j < 4; ++j) {
        int n2 = nx * 32 + ty + j * 8;                 // output row (column of W_out)
        int k2 = ky * 32 + tx;                         // output col (row of W_out)
        Wt[(long)n2 * H + k2] = f2bf(tile[tx][ty + j * 8]);
    }
}

// ---------------- one-time: memories = concat(tree,enc) @ W_enc + b_enc ------------
__global__ void memories_kernel(const float* __restrict__ tree_output,
                                const float* __restrict__ encoder_outputs,
                                const float* __restrict__ W_enc,
                                const float* __restrict__ b_enc,
                                float* __restrict__ memories) {
    __shared__ float rows[8][2 * H];                   // 8 rows of 1024 f32 = 32 KB
    int blk = blockIdx.x, tid = threadIdx.x;           // 512 threads
    for (int i = tid; i < 8 * 2 * H; i += 512) {
        int r = i >> 10, k = i & 1023;
        int g = blk * 8 + r;
        int bb = g / S, ss = g % S;
        rows[r][k] = (k < H) ? tree_output[((long)bb * S + ss) * H + k]
                             : encoder_outputs[((long)bb * S + ss) * H + (k - H)];
    }
    __syncthreads();
    float acc[8];
    float bias = b_enc[tid];
    #pragma unroll
    for (int r = 0; r < 8; ++r) acc[r] = bias;
    for (int k = 0; k < 2 * H; ++k) {
        float w = W_enc[k * H + tid];                  // coalesced across tid
        #pragma unroll
        for (int r = 0; r < 8; ++r) acc[r] += rows[r][k] * w;
    }
    #pragma unroll
    for (int r = 0; r < 8; ++r)
        memories[(long)(blk * 8 + r) * H + tid] = acc[r];
}

// ---------------- one-time: state init --------------------------------------------
__global__ void init_state_kernel(const float* __restrict__ h0, const float* __restrict__ c0,
                                  float* __restrict__ h, float* __restrict__ c,
                                  float* __restrict__ ctx) {
    int i = blockIdx.x * blockDim.x + threadIdx.x;
    if (i < B * H) { h[i] = h0[i]; c[i] = c0[i]; ctx[i] = 0.0f; }
}

// ---------------- per-step: x = [emb(y); ctx] @ W_red + b_red ----------------------
__global__ void x_kernel(const int* __restrict__ trg_seq,
                         const float* __restrict__ embedding,
                         const float* __restrict__ ctx_cur,
                         const float* __restrict__ W_red, const float* __restrict__ b_red,
                         float* __restrict__ x, int t) {
    __shared__ float cat[E + H];                       // 812
    int b = blockIdx.x, tid = threadIdx.x;             // 512 threads
    int y = trg_seq[b * T + t];
    for (int i = tid; i < E + H; i += 512)
        cat[i] = (i < E) ? embedding[(long)y * E + i] : ctx_cur[b * H + (i - E)];
    __syncthreads();
    if (tid < E) {
        float acc = b_red[tid];
        for (int k = 0; k < E + H; ++k) acc += cat[k] * W_red[k * E + tid];
        x[b * E + tid] = acc;
    }
}

// ---------------- per-step: gates GEMM + LSTM pointwise (fused per row) ------------
__global__ void __launch_bounds__(1024)
gates_lstm_kernel(const float* __restrict__ x,
                  const float* __restrict__ h_cur, const float* __restrict__ c_cur,
                  const float* __restrict__ W_ih, const float* __restrict__ W_hh,
                  const float* __restrict__ b_ih, const float* __restrict__ b_hh,
                  float* __restrict__ h_nxt, float* __restrict__ c_nxt) {
    __shared__ float xh[E + H];                        // 812
    __shared__ float g[4 * H];                         // 2048
    int b = blockIdx.x, tid = threadIdx.x;             // 1024 threads
    for (int i = tid; i < E + H; i += 1024)
        xh[i] = (i < E) ? x[b * E + i] : h_cur[b * H + (i - E)];
    __syncthreads();
    int n0 = tid, n1 = tid + 1024;
    float a0 = b_ih[n0] + b_hh[n0];
    float a1 = b_ih[n1] + b_hh[n1];
    for (int k = 0; k < E; ++k) {
        float v = xh[k];
        a0 += v * W_ih[k * (4 * H) + n0];
        a1 += v * W_ih[k * (4 * H) + n1];
    }
    for (int k = 0; k < H; ++k) {
        float v = xh[E + k];
        a0 += v * W_hh[k * (4 * H) + n0];
        a1 += v * W_hh[k * (4 * H) + n1];
    }
    g[n0] = a0; g[n1] = a1;
    __syncthreads();
    if (tid < H) {                                     // i,f,g,o torch gate order
        float ig = g[tid], fg = g[H + tid], gg = g[2 * H + tid], og = g[3 * H + tid];
        float si = 1.0f / (1.0f + __expf(-ig));
        float sf = 1.0f / (1.0f + __expf(-fg));
        float so = 1.0f / (1.0f + __expf(-og));
        float cn = sf * c_cur[b * H + tid] + si * tanhf(gg);
        c_nxt[b * H + tid] = cn;
        h_nxt[b * H + tid] = so * tanhf(cn);
    }
}

// ---------------- per-step: energy[b,s] = h . memories[b,s] (masked) ---------------
__global__ void energy_kernel(const float* __restrict__ h_nxt,
                              const float* __restrict__ memories,
                              const unsigned char* __restrict__ mask,
                              float* __restrict__ energy) {
    int s = blockIdx.x, b = blockIdx.y, tid = threadIdx.x;   // 64 threads
    const float* hv = h_nxt + b * H;
    const float* mv = memories + ((long)b * S + s) * H;
    float p = 0.0f;
    for (int k = tid; k < H; k += 64) p += hv[k] * mv[k];
    __shared__ float red[64];
    red[tid] = p; __syncthreads();
    for (int o = 32; o > 0; o >>= 1) { if (tid < o) red[tid] += red[tid + o]; __syncthreads(); }
    if (tid == 0) energy[b * S + s] = mask[b * S + s] ? NEG_INF : red[0];
}

// ---------------- per-step: softmax over S + ctx = attn @ memories -----------------
__global__ void attn_ctx_kernel(const float* __restrict__ energy,
                                const float* __restrict__ memories,
                                float* __restrict__ ctx_nxt) {
    __shared__ float e[S];
    __shared__ float inv_s;
    int b = blockIdx.x, tid = threadIdx.x;             // 512 threads
    if (tid < S) e[tid] = energy[b * S + tid];
    __syncthreads();
    if (tid == 0) {
        float mx = -3.4e38f;
        for (int s = 0; s < S; ++s) mx = fmaxf(mx, e[s]);
        float sm = 0.0f;
        for (int s = 0; s < S; ++s) { e[s] = __expf(e[s] - mx); sm += e[s]; }
        inv_s = 1.0f / sm;
    }
    __syncthreads();
    float acc = 0.0f;
    for (int s = 0; s < S; ++s) acc += e[s] * memories[((long)b * S + s) * H + tid];
    ctx_nxt[b * H + tid] = acc * inv_s;
}

// ---------------- per-step: z = tanh([h;ctx] @ W_cat + b_cat), stored bf16 ---------
__global__ void z_kernel(const float* __restrict__ h_nxt, const float* __restrict__ ctx_nxt,
                         const float* __restrict__ W_cat, const float* __restrict__ b_cat,
                         unsigned short* __restrict__ z_bf) {
    __shared__ float cat[2 * H];
    int b = blockIdx.x, tid = threadIdx.x;             // 512 threads
    cat[tid]     = h_nxt[b * H + tid];
    cat[H + tid] = ctx_nxt[b * H + tid];
    __syncthreads();
    float acc = b_cat[tid];
    for (int k = 0; k < 2 * H; ++k) acc += cat[k] * W_cat[k * H + tid];
    z_bf[b * H + tid] = f2bf(tanhf(acc));
}

// ---------------- per-step: logits = z @ W_out + b_out  (WMMA bf16 -> f32) ---------
// Block: 256 threads = 8 wave32 waves. Each wave owns one 16x16 tile of the 32x64
// block tile: mt = wave&1 (two 16-row M tiles of B=32), nt = wave>>1 (four N tiles).
// K loop: 512 / 32 = 16 x v_wmma_f32_16x16x32_bf16.
__global__ void __launch_bounds__(256)
logits_wmma_kernel(const unsigned short* __restrict__ Wt,   // [NPAD, H] bf16
                   const unsigned short* __restrict__ z_bf, // [B, H]    bf16
                   const float* __restrict__ b_out,
                   float* __restrict__ out, int t) {
    int tid  = threadIdx.x;
    int wave = tid >> 5;
    int lane = tid & 31;
    int mt = wave & 1;
    int nt = wave >> 1;
    int n0 = blockIdx.x * 64 + nt * 16;

    // A fragment (16x32 bf16, ISA layout): lane<16 -> K {k0+0..7, k0+16..23},
    // lane>=16 -> K {k0+8..15, k0+24..31}; row M = mt*16 + (lane&15).
    int m = mt * 16 + (lane & 15);
    const unsigned short* zrow = z_bf + m * H;
    int aoff0 = (lane < 16) ? 0 : 8;
    int aoff1 = aoff0 + 16;

    // B fragment (32x16 bf16): column N = n0 + (lane&15); lane<16 holds K {k0+0..7, k0+16..23},
    // lane>=16 holds K {k0+8..15, k0+24..31} (same K order as A).
    const unsigned short* wrow = Wt + (long)(n0 + (lane & 15)) * H + ((lane < 16) ? 0 : 8);

    v8f acc = {};
    #pragma unroll
    for (int k0 = 0; k0 < H; k0 += 32) {
        Frag A, Bf;
        A.q[0]  = *(const u32x4*)(zrow + k0 + aoff0);
        A.q[1]  = *(const u32x4*)(zrow + k0 + aoff1);
        Bf.q[0] = *(const u32x4*)(wrow + k0);
        Bf.q[1] = *(const u32x4*)(wrow + k0 + 16);
        acc = __builtin_amdgcn_wmma_f32_16x16x32_bf16(
            false, A.v, false, Bf.v, (short)0, acc, false, false);
    }

    // C/D layout: lane<16 -> N=lane, VGPR r -> M=r; lane>=16 -> N=lane-16, M=8+r.
    int n = n0 + (lane & 15);
    int mbase = mt * 16 + ((lane < 16) ? 0 : 8);
    if (n < VEXT) {
        float bias = (n < V) ? b_out[n] : 0.0f;
        #pragma unroll
        for (int r = 0; r < 8; ++r) {
            int b = mbase + r;
            float vout = (n < V) ? (acc[r] + bias) : 0.0f;   // OOV region = 0 (ext)
            out[((long)b * T + t) * VEXT + n] = vout;
        }
    }
}

// ---------------- per-step: pointer scatter-max into extended vocab ----------------
__global__ void scatter_kernel(const int* __restrict__ ext_src_seq,
                               const float* __restrict__ energy,
                               float* __restrict__ out, int t) {
    __shared__ float e[S];
    __shared__ int  id[S];
    int b = blockIdx.x, tid = threadIdx.x;             // 128 threads
    if (tid < S) { e[tid] = energy[b * S + tid]; id[tid] = ext_src_seq[b * S + tid]; }
    __syncthreads();
    for (int s = tid; s < S; s += 128) {
        int v = id[s];
        float m = -3.4e38f;
        bool first = true;
        for (int s2 = 0; s2 < S; ++s2) {
            if (id[s2] == v) { m = fmaxf(m, e[s2]); if (s2 < s) first = false; }
        }
        // scat == -INF (all duplicates masked) maps to 0 in the reference -> no-op
        if (first && m != NEG_INF) {
            float* p = out + ((long)b * T + t) * VEXT + v;   // unique writer per index
            *p = *p + m;
        }
    }
}

// ---------------- per-step: out == 0.0 -> -INF -------------------------------------
__global__ void finalize_kernel(float* __restrict__ out, int t) {
    int i = blockIdx.x * blockDim.x + threadIdx.x;
    if (i >= B * VEXT) return;
    int b = i / VEXT, n = i % VEXT;
    long p = ((long)b * T + t) * VEXT + n;
    if (out[p] == 0.0f) out[p] = NEG_INF;
}

// ===================================================================================
extern "C" void kernel_launch(void* const* d_in, const int* in_sizes, int n_in,
                              void* d_out, int out_size, void* d_ws, size_t ws_size,
                              hipStream_t stream) {
    (void)in_sizes; (void)n_in; (void)out_size; (void)ws_size;
    const int*            trg_seq         = (const int*)d_in[0];
    const int*            ext_src_seq     = (const int*)d_in[1];
    const float*          h0              = (const float*)d_in[2];
    const float*          c0              = (const float*)d_in[3];
    const float*          tree_output     = (const float*)d_in[4];
    /* d_in[5] tree_enc_states: unused by reference */
    const float*          encoder_outputs = (const float*)d_in[6];
    const unsigned char*  encoder_mask    = (const unsigned char*)d_in[7];
    const float*          embedding       = (const float*)d_in[8];
    const float*          W_enc  = (const float*)d_in[9];
    const float*          b_enc  = (const float*)d_in[10];
    const float*          W_red  = (const float*)d_in[11];
    const float*          b_red  = (const float*)d_in[12];
    const float*          W_ih   = (const float*)d_in[13];
    const float*          W_hh   = (const float*)d_in[14];
    const float*          b_ih   = (const float*)d_in[15];
    const float*          b_hh   = (const float*)d_in[16];
    const float*          W_cat  = (const float*)d_in[17];
    const float*          b_cat  = (const float*)d_in[18];
    const float*          W_out  = (const float*)d_in[19];
    const float*          b_out  = (const float*)d_in[20];
    float* out = (float*)d_out;

    // ---- workspace carving (~53 MB) ----
    char* ws = (char*)d_ws;
    size_t off = 0;
    auto take = [&](size_t bytes) { char* p = ws + off; off = (off + bytes + 255) & ~(size_t)255; return p; };
    unsigned short* Wt       = (unsigned short*)take((size_t)NPAD * H * 2);  // 46.1 MB
    float*          memories = (float*)take((size_t)B * S * H * 4);          //  6.6 MB
    float*          hbuf     = (float*)take((size_t)2 * B * H * 4);
    float*          cbuf     = (float*)take((size_t)2 * B * H * 4);
    float*          ctxbuf   = (float*)take((size_t)2 * B * H * 4);
    float*          xbuf     = (float*)take((size_t)B * E * 4);
    float*          energy   = (float*)take((size_t)B * S * 4);
    unsigned short* z_bf     = (unsigned short*)take((size_t)B * H * 2);

    // ---- one-time precompute ----
    wt_transpose_kernel<<<dim3(NPAD / 32, H / 32), dim3(32, 8), 0, stream>>>(W_out, Wt);
    memories_kernel<<<(B * S) / 8, 512, 0, stream>>>(tree_output, encoder_outputs,
                                                     W_enc, b_enc, memories);
    init_state_kernel<<<(B * H + 255) / 256, 256, 0, stream>>>(h0, c0, hbuf, cbuf, ctxbuf);

    // ---- sequential decode, state ping-pong ----
    for (int t = 0; t < T; ++t) {
        float* h_cur   = hbuf   + (t & 1) * (B * H);
        float* h_nxt   = hbuf   + ((t + 1) & 1) * (B * H);
        float* c_cur   = cbuf   + (t & 1) * (B * H);
        float* c_nxt   = cbuf   + ((t + 1) & 1) * (B * H);
        float* ctx_cur = ctxbuf + (t & 1) * (B * H);
        float* ctx_nxt = ctxbuf + ((t + 1) & 1) * (B * H);

        x_kernel<<<B, 512, 0, stream>>>(trg_seq, embedding, ctx_cur, W_red, b_red, xbuf, t);
        gates_lstm_kernel<<<B, 1024, 0, stream>>>(xbuf, h_cur, c_cur, W_ih, W_hh,
                                                  b_ih, b_hh, h_nxt, c_nxt);
        energy_kernel<<<dim3(S, B), 64, 0, stream>>>(h_nxt, memories, encoder_mask, energy);
        attn_ctx_kernel<<<B, 512, 0, stream>>>(energy, memories, ctx_nxt);
        z_kernel<<<B, 512, 0, stream>>>(h_nxt, ctx_nxt, W_cat, b_cat, z_bf);
        logits_wmma_kernel<<<NPAD / 64, 256, 0, stream>>>(Wt, z_bf, b_out, out, t);
        scatter_kernel<<<B, 128, 0, stream>>>(ext_src_seq, energy, out, t);
        finalize_kernel<<<(B * VEXT + 255) / 256, 256, 0, stream>>>(out, t);
    }
}
